// NeighborMLPConvLayer_15350213116605
// MI455X (gfx1250) — compile-verified, matrix operand order checked by reading, not run on hardware
//
#include <hip/hip_runtime.h>
#include <hip/hip_bf16.h>
#include <math.h>

typedef __attribute__((ext_vector_type(16))) __bf16 v16bf;
typedef __attribute__((ext_vector_type(8)))  float  v8f;

#define C_IN 32
#define HID  128
#define OUT  64

__device__ __forceinline__ unsigned short f2bf(float f) {
  unsigned int u = __float_as_uint(f);
  u += 0x7FFFu + ((u >> 16) & 1u);            // round-to-nearest-even
  return (unsigned short)(u >> 16);
}

// Branch-free GELU (tanh form reduced to sigmoid, log2e folded into constants):
//   gelu(x) = x * sigmoid(1.5957691*x + 0.0713548*x^3)
//           = x * rcp(1 + exp2(-x*(2.3023467 + 0.1029445*x^2)))
// 5 VALU + 2 TRANS (v_exp_f32, v_rcp_f32), saturates correctly at +/-inf.
__device__ __forceinline__ float gelu_fast(float x) {
  float x2 = x * x;
  float w  = x * __builtin_fmaf(x2, -0.10294455f, -2.30234666f);
  float e  = __builtin_amdgcn_exp2f(w);
  return x * __builtin_amdgcn_rcpf(1.0f + e);
}

union Frag {
  v16bf v;
  unsigned int u[8];
};

// K-pair offset inside a 16x32 (or 32x16) bf16 fragment, per ISA 7.12.2 layout:
// VGPR j<4 : K = g*8 + 2j ; VGPR j>=4 : K = 16 + g*8 + 2(j-4)   (g = lane/16)
__device__ __forceinline__ int koff(int j, int g) {
  return (j < 4) ? (g * 8 + 2 * j) : (16 + g * 8 + 2 * (j - 4));
}

// ---------------------------------------------------------------- prep kernels
__global__ void zero_accum(float* __restrict__ acc, int n) {
  int t = blockIdx.x * blockDim.x + threadIdx.x;
  if (t < n) acc[t] = 0.0f;
}

__global__ void build_segids(const int* __restrict__ rs, int* __restrict__ seg, int M_) {
  int s = blockIdx.x * blockDim.x + threadIdx.x;
  if (s >= M_) return;
  int lo = rs[s], hi = rs[s + 1];
  for (int e = lo; e < hi; ++e) seg[e] = s;
}

// W1 [64][128] row-major -> W1t bf16 [128][64]  (N-major, K contiguous)
// W2 [128][64] row-major -> W2t bf16 [64][128]
__global__ void prep_weights(const float* __restrict__ W1, const float* __restrict__ W2,
                             unsigned short* __restrict__ W1t, unsigned short* __restrict__ W2t) {
  int t = blockIdx.x * blockDim.x + threadIdx.x;
  if (t < 64 * 128) {
    int n1 = t >> 6, k1 = t & 63;                 // n<128, k<64
    W1t[n1 * 64 + k1] = f2bf(W1[k1 * 128 + n1]);
    int n2 = t >> 7, k2 = t & 127;                // n<64, k<128
    W2t[n2 * 128 + k2] = f2bf(W2[k2 * 64 + n2]);
  }
}

// ---------------------------------------------------------------- main kernel
__global__ void __launch_bounds__(128) mlp_main(
    const float* __restrict__ inF, const float* __restrict__ outF,
    const unsigned short* __restrict__ gW1t, const unsigned short* __restrict__ gW2t,
    const float* __restrict__ b1,
    const int* __restrict__ nbrIdx, const int* __restrict__ segIds,
    float* __restrict__ accum, int E_, int numTiles)
{
  __shared__ unsigned short sW1t[HID * 64];        // 16 KB  [n][k], n<128 k<64
  __shared__ unsigned short sW2t[OUT * HID];       // 16 KB  [n][k], n<64  k<128
  __shared__ unsigned short sStage[4][16 * HID];   // 4 x 4 KB per-wave staging

  // cooperative weight load (dword copies)
  {
    const unsigned int* s1 = (const unsigned int*)gW1t;
    const unsigned int* s2 = (const unsigned int*)gW2t;
    unsigned int* d1 = (unsigned int*)sW1t;
    unsigned int* d2 = (unsigned int*)sW2t;
    for (int i = threadIdx.x; i < (HID * 64) / 2; i += 128) d1[i] = s1[i];
    for (int i = threadIdx.x; i < (OUT * HID) / 2; i += 128) d2[i] = s2[i];
  }
  __syncthreads();

  const int wave = threadIdx.x >> 5;
  const int lane = threadIdx.x & 31;
  const int g = lane >> 4;       // lane half
  const int m = lane & 15;       // A-row / B-C column index for this lane
  unsigned short* stage = sStage[wave];

  // tile-invariant per-lane b1 biases (column n = nt*16 + m)
  float bias1[8];
  #pragma unroll
  for (int nt = 0; nt < 8; ++nt) bias1[nt] = b1[nt * 16 + m];

  for (int tile = blockIdx.x * 4 + wave; tile < numTiles; tile += gridDim.x * 4) {
    const int base = tile * 16;

    // ---- gather agg tile (16 x 64) -> bf16, row-major in stage -------------
    {
      int e = base + m;                 // lanes 0-15: neighbor half, 16-31: self half
      if (e < E_) {
        int row = (g == 0) ? nbrIdx[e] : segIds[e];
        const float* src = ((g == 0) ? inF : outF) + (long long)row * C_IN;
        #pragma unroll
        for (int i = 0; i < C_IN; i += 4) {
          float4 f = *(const float4*)(src + i);
          unsigned int p0 = (unsigned int)f2bf(f.x) | ((unsigned int)f2bf(f.y) << 16);
          unsigned int p1 = (unsigned int)f2bf(f.z) | ((unsigned int)f2bf(f.w) << 16);
          unsigned int* dst = (unsigned int*)&stage[m * 64 + g * 32 + i];
          dst[0] = p0; dst[1] = p1;
        }
      } else {
        #pragma unroll
        for (int i = 0; i < C_IN; i += 2)
          *(unsigned int*)&stage[m * 64 + g * 32 + i] = 0u;
      }
    }
    __asm__ volatile("" ::: "memory");   // order LDS staging stores vs. fragment loads

    // ---- A fragments (16x32 bf16 x2) ---------------------------------------
    Frag a0, a1;
    #pragma unroll
    for (int j = 0; j < 8; ++j) {
      a0.u[j] = *(const unsigned int*)&stage[m * 64 +  0 + koff(j, g)];
      a1.u[j] = *(const unsigned int*)&stage[m * 64 + 32 + koff(j, g)];
    }

    // ---- MLP1: agg @ W1 + b1, fast GELU, restage h (16x128 bf16) -----------
    #pragma unroll
    for (int nt = 0; nt < 8; ++nt) {
      const int n = nt * 16 + m;        // output column for this lane
      v8f acc;
      #pragma unroll
      for (int r = 0; r < 8; ++r) acc[r] = bias1[nt];  // fuse bias into accumulator

      Frag bf0, bf1;
      #pragma unroll
      for (int j = 0; j < 8; ++j) {
        bf0.u[j] = *(const unsigned int*)&sW1t[n * 64 +  0 + koff(j, g)];
        bf1.u[j] = *(const unsigned int*)&sW1t[n * 64 + 32 + koff(j, g)];
      }
      acc = __builtin_amdgcn_wmma_f32_16x16x32_bf16(false, a0.v, false, bf0.v,
                                                    (short)0, acc, false, false);
      acc = __builtin_amdgcn_wmma_f32_16x16x32_bf16(false, a1.v, false, bf1.v,
                                                    (short)0, acc, false, false);
      #pragma unroll
      for (int r = 0; r < 8; ++r) {
        int mr = r + 8 * g;             // C-matrix row held by this lane/VGPR
        stage[mr * HID + n] = f2bf(gelu_fast(acc[r]));
      }
    }
    __asm__ volatile("" ::: "memory");   // order h stores vs. MLP2 fragment loads

    // ---- MLP2: h (16x128) @ W2 (128x64) ------------------------------------
    v8f acc2[4];
    #pragma unroll
    for (int nt = 0; nt < 4; ++nt)
      #pragma unroll
      for (int r = 0; r < 8; ++r) acc2[nt][r] = 0.0f;

    #pragma unroll
    for (int kb = 0; kb < 4; ++kb) {
      Frag ha;
      #pragma unroll
      for (int j = 0; j < 8; ++j)
        ha.u[j] = *(const unsigned int*)&stage[m * HID + kb * 32 + koff(j, g)];
      #pragma unroll
      for (int nt = 0; nt < 4; ++nt) {
        const int n = nt * 16 + m;
        Frag wb;
        #pragma unroll
        for (int j = 0; j < 8; ++j)
          wb.u[j] = *(const unsigned int*)&sW2t[n * HID + kb * 32 + koff(j, g)];
        acc2[nt] = __builtin_amdgcn_wmma_f32_16x16x32_bf16(false, ha.v, false, wb.v,
                                                           (short)0, acc2[nt], false, false);
      }
    }

    // ---- scatter-add into per-segment f32 accumulator ----------------------
    #pragma unroll
    for (int r = 0; r < 8; ++r) {
      int mr = r + 8 * g;
      int e = base + mr;
      if (e < E_) {
        int seg = segIds[e];
        float* dst = accum + (long long)seg * OUT + m;
        #pragma unroll
        for (int nt = 0; nt < 4; ++nt)
          atomicAdd(dst + nt * 16, acc2[nt][r]);
      }
    }
    __asm__ volatile("" ::: "memory");
  }
}

// ---------------------------------------------------------------- finalize
__global__ void finalize(const float* __restrict__ accum, const int* __restrict__ rs,
                         const float* __restrict__ b2, float* __restrict__ out, int M_) {
  int t = blockIdx.x * blockDim.x + threadIdx.x;
  if (t >= M_ * OUT) return;
  int s = t >> 6;
  int o = t & 63;
  int cnt = rs[s + 1] - rs[s];
  out[t] = (cnt > 0) ? (accum[t] / (float)cnt + b2[o]) : 0.0f;
}

// ---------------------------------------------------------------- launch
extern "C" void kernel_launch(void* const* d_in, const int* in_sizes, int n_in,
                              void* d_out, int out_size, void* d_ws, size_t ws_size,
                              hipStream_t stream) {
  const float* inF  = (const float*)d_in[0];
  const float* outF = (const float*)d_in[1];
  const float* W1   = (const float*)d_in[2];
  const float* b1   = (const float*)d_in[3];
  const float* W2   = (const float*)d_in[4];
  const float* b2   = (const float*)d_in[5];
  const int* nbrIdx = (const int*)d_in[6];
  const int* rs     = (const int*)d_in[7];

  const int E_ = in_sizes[6];
  const int M_ = in_sizes[7] - 1;

  char* ws = (char*)d_ws;
  size_t off = 0;
  float* accum = (float*)(ws + off);               off += (size_t)M_ * OUT * sizeof(float);
  int* segIds = (int*)(ws + off);                  off += (size_t)E_ * sizeof(int);
  unsigned short* W1t = (unsigned short*)(ws + off); off += (size_t)64 * 128 * sizeof(unsigned short);
  unsigned short* W2t = (unsigned short*)(ws + off); off += (size_t)128 * 64 * sizeof(unsigned short);
  (void)off; (void)ws_size; (void)n_in; (void)out_size;

  const int nAcc = M_ * OUT;
  zero_accum<<<(nAcc + 255) / 256, 256, 0, stream>>>(accum, nAcc);
  build_segids<<<(M_ + 255) / 256, 256, 0, stream>>>(rs, segIds, M_);
  prep_weights<<<(64 * 128 + 255) / 256, 256, 0, stream>>>(W1, W2, W1t, W2t);

  const int numTiles = (E_ + 15) / 16;
  int blocks = (numTiles + 3) / 4;
  if (blocks > 8192) blocks = 8192;
  mlp_main<<<blocks, 128, 0, stream>>>(inF, outF, W1t, W2t, b1, nbrIdx, segIds,
                                       accum, E_, numTiles);

  finalize<<<(nAcc + 255) / 256, 256, 0, stream>>>(accum, rs, b2, (float*)d_out, M_);
}
